// RNN_71236327571777
// MI455X (gfx1250) — matrix operand order repaired and checked by reading, not compiled, not run on hardware
//
#include <hip/hip_runtime.h>
#include <cmath>
#include <type_traits>

typedef __attribute__((ext_vector_type(16))) __bf16 v16bf;
typedef __attribute__((ext_vector_type(8)))  __bf16 v8bf;
typedef __attribute__((ext_vector_type(8)))  float  v8f;

#define HID 512
#define BATCH 64
#define SEQ 1024

static __device__ inline v16bf bf_cat(v8bf lo, v8bf hi) {
  return __builtin_shufflevector(lo, hi, 0,1,2,3,4,5,6,7,8,9,10,11,12,13,14,15);
}
static __device__ inline v8bf cvt8(float4 x, float4 y) {
  v8bf o = { (__bf16)x.x, (__bf16)x.y, (__bf16)x.z, (__bf16)x.w,
             (__bf16)y.x, (__bf16)y.y, (__bf16)y.z, (__bf16)y.w };
  return o;
}

// ---------------------------------------------------------------------------
// Prep: bf16-transpose w_xh and w_hh, bf16-copy w_out (already [out,in] = B^T
// layout), zero bf16 h0 and the grid-barrier words. Runs every call.
// ---------------------------------------------------------------------------
__global__ __launch_bounds__(256) void rnn_prep(
    const float* __restrict__ w_xh, const float* __restrict__ w_hh,
    const float* __restrict__ w_out,
    __bf16* __restrict__ wxhT, __bf16* __restrict__ whhT,
    __bf16* __restrict__ woutB, __bf16* __restrict__ h0,
    unsigned* __restrict__ bar) {
  int i = blockIdx.x * 256 + threadIdx.x;          // 0 .. 512*512-1
  int r = i >> 9, c = i & 511;
  wxhT[i]  = (__bf16)w_xh[c * HID + r];            // wxhT[h][i] = w_xh[i][h]
  whhT[i]  = (__bf16)w_hh[c * HID + r];            // whhT[n][k] = w_hh[k][n]
  woutB[i] = (__bf16)w_out[i];                     // already [o][k]
  if (i < BATCH * HID) h0[i] = (__bf16)0.0f;
  if (i < 2) bar[i] = 0u;
}

// ---------------------------------------------------------------------------
// Generic 64x64-block GEMM:  out = A * BT(bf16,[N][K]) + bias,  K = 512.
// TA = float (phase A: converts to bf16 while staging to LDS) or __bf16
// (phase C: h_final already bf16).  8 waves = 4 Mtiles x 2 Ngroups, each wave
// owns 2 adjacent 16x16 N tiles; B fragments are contiguous 16B global loads.
// ---------------------------------------------------------------------------
template <typename TA>
__global__ __launch_bounds__(256) void gemm64_bias(
    const TA* __restrict__ A, long a_blk_stride, long a_rstride,
    const __bf16* __restrict__ BT, const float* __restrict__ bias,
    float* __restrict__ out, long o_blk_stride) {
  __shared__ __bf16 sA[64 * 32];

  const int tid  = threadIdx.x;
  const int lane = tid & 31;
  const int wv   = tid >> 5;            // 0..7
  const int mt   = wv & 3;              // M tile 0..3
  const int ng   = wv >> 2;             // N group 0..1
  const int half = lane >> 4;           // 0/1
  const int l16  = lane & 15;

  const long a_base = (long)blockIdx.x * a_blk_stride;
  const int  arow   = mt * 16 + l16;    // A row within 64-row block
  const int  akb    = half * 16;        // K sub-offset {0,16}
  const int  nc0    = blockIdx.y * 64 + ng * 32 + l16;      // global col tile0
  const int  nc1    = nc0 + 16;                             // tile1

  v8f acc0 = {}; v8f acc1 = {};

  for (int kc = 0; kc < 16; ++kc) {     // K chunks of 32
    // stage A[64][32] -> bf16 LDS (8 elems / thread)
    {
      int e  = tid * 8;
      int r  = e >> 5, c0 = e & 31;
      const TA* src = A + a_base + (long)r * a_rstride + kc * 32 + c0;
      if constexpr (std::is_same<TA, float>::value) {
        float4 x = ((const float4*)src)[0];
        float4 y = ((const float4*)src)[1];
        *(v8bf*)&sA[e] = cvt8(x, y);
      } else {
        *(v8bf*)&sA[e] = *(const v8bf*)src;
      }
    }
    __syncthreads();

    v16bf a = bf_cat(*(const v8bf*)&sA[arow * 32 + akb],
                     *(const v8bf*)&sA[arow * 32 + akb + 8]);
    const __bf16* bp0 = BT + (long)nc0 * HID + kc * 32 + half * 16;
    const __bf16* bp1 = BT + (long)nc1 * HID + kc * 32 + half * 16;
    v16bf b0 = bf_cat(((const v8bf*)bp0)[0], ((const v8bf*)bp0)[1]);
    v16bf b1 = bf_cat(((const v8bf*)bp1)[0], ((const v8bf*)bp1)[1]);

    acc0 = __builtin_amdgcn_wmma_f32_16x16x32_bf16(false, a, false, b0,
                                                   (short)0, acc0, false, false);
    acc1 = __builtin_amdgcn_wmma_f32_16x16x32_bf16(false, a, false, b1,
                                                   (short)0, acc1, false, false);
    __syncthreads();
  }

  const long obase = (long)blockIdx.x * o_blk_stride;
  const float bs0 = bias[nc0], bs1 = bias[nc1];
#pragma unroll
  for (int j = 0; j < 8; ++j) {
    int m = mt * 16 + j + half * 8;     // D layout: VGPR j -> row j+8*half
    out[obase + (long)m * HID + nc0] = acc0[j] + bs0;
    out[obase + (long)m * HID + nc1] = acc1[j] + bs1;
  }
}

// ---------------------------------------------------------------------------
// Persistent recurrence: 16 WGs x 256 threads, 1024 serial steps.
// h lives in global as bf16 (double-buffered, 64KB each); each step every WG
// pulls the full h into LDS with GLOBAL_LOAD_ASYNC_TO_LDS_B128 (ASYNCcnt),
// runs one 16x16x(K=512) WMMA tile per wave (two independent accumulator
// chains to halve WMMA RAW latency) against its pinned w_hh^T slice, applies
// tanh(acc + xw_t), stores bf16 h_next, then crosses a device-scope atomic
// barrier.  LDS: 64KB h + 32KB weights (CDNA5 320KB/WGP).
// ---------------------------------------------------------------------------
__global__ __launch_bounds__(256) void rnn_recurrence(
    const float* __restrict__ xw,        // [T][B][H] f32 (bias included)
    const __bf16* __restrict__ whhT,     // [n][k] bf16
    __bf16* __restrict__ h0, __bf16* __restrict__ h1,
    unsigned* __restrict__ bar) {
  __shared__ __bf16 sH[BATCH * HID];     // 64 KB
  __shared__ __bf16 sW[32 * HID];        // 32 KB

  const int tid = threadIdx.x;
  const int wg  = blockIdx.x;            // 0..15

  // pin w_hh^T slice for our 32 columns
  {
    int j  = tid >> 3;                   // 0..31
    int k0 = (tid & 7) * 64;
    const v8bf* src = (const v8bf*)(whhT + (long)(wg * 32 + j) * HID + k0);
    v8bf* dst = (v8bf*)&sW[j * HID + k0];
#pragma unroll
    for (int i = 0; i < 8; ++i) dst[i] = src[i];
  }

  const int lane = tid & 31;
  const int wv   = tid >> 5;
  const int mt   = wv & 3;               // batch tile 0..3
  const int nt   = wv >> 2;              // col tile 0..1
  const int half = lane >> 4;
  const int l16  = lane & 15;
  const int arow = mt * 16 + l16;
  const int akb  = half * 16;
  const int bcol = nt * 16 + l16;        // within 32-col slice
  const int c    = wg * 32 + nt * 16 + l16;   // global output column

  const unsigned sH_base = (unsigned)(uintptr_t)(void*)&sH[0]; // LDS byte offset

  __bf16* hb[2] = { h0, h1 };

  for (int t = 0; t < SEQ; ++t) {
    const __bf16* hc = hb[t & 1];
    __bf16* hn       = hb[(t + 1) & 1];

    // async-copy h (64KB bf16) global -> LDS: 16 x b128 per thread
    {
      const unsigned lds0 = sH_base + (unsigned)tid * 256u;
      const char* g0 = (const char*)hc + tid * 256;
#pragma unroll
      for (int i = 0; i < 16; ++i) {
        unsigned l = lds0 + (unsigned)(i * 16);
        unsigned long long g = (unsigned long long)(uintptr_t)(g0 + i * 16);
        asm volatile("global_load_async_to_lds_b128 %0, %1, off"
                     :: "v"(l), "v"(g) : "memory");
      }
      asm volatile("s_wait_asynccnt 0x0" ::: "memory");
    }
    __syncthreads();

    // prefetch next step's xw lines into WGP-level cache (locality 3 -> WGP
    // scope: pulls into all cache levels; this WGP reads them next step)
    if (t + 1 < SEQ) {
      const char* nxt = (const char*)(xw + (long)(t + 1) * (BATCH * HID));
#pragma unroll
      for (int j = 0; j < 8; ++j) {
        int b = mt * 16 + j + half * 8;
        __builtin_prefetch(nxt + ((long)b * HID + c) * 4, 0, 3);
      }
    }

    // two independent WMMA accumulator chains (even/odd K chunks)
    v8f accA = {}; v8f accB = {};
#pragma unroll
    for (int kc = 0; kc < 16; kc += 2) {
      v16bf a0 = bf_cat(*(const v8bf*)&sH[arow * HID + kc * 32 + akb],
                        *(const v8bf*)&sH[arow * HID + kc * 32 + akb + 8]);
      v16bf b0 = bf_cat(*(const v8bf*)&sW[bcol * HID + kc * 32 + half * 16],
                        *(const v8bf*)&sW[bcol * HID + kc * 32 + half * 16 + 8]);
      v16bf a1 = bf_cat(*(const v8bf*)&sH[arow * HID + (kc + 1) * 32 + akb],
                        *(const v8bf*)&sH[arow * HID + (kc + 1) * 32 + akb + 8]);
      v16bf b1 = bf_cat(*(const v8bf*)&sW[bcol * HID + (kc + 1) * 32 + half * 16],
                        *(const v8bf*)&sW[bcol * HID + (kc + 1) * 32 + half * 16 + 8]);
      accA = __builtin_amdgcn_wmma_f32_16x16x32_bf16(false, a0, false, b0,
                                                     (short)0, accA, false, false);
      accB = __builtin_amdgcn_wmma_f32_16x16x32_bf16(false, a1, false, b1,
                                                     (short)0, accB, false, false);
    }

    // epilogue: tanh(accA + accB + xw_t) -> bf16 h_next
    const float* xwt = xw + (long)t * (BATCH * HID);
#pragma unroll
    for (int j = 0; j < 8; ++j) {
      int b = mt * 16 + j + half * 8;
      float v = accA[j] + accB[j] + xwt[(long)b * HID + c];
      hn[(long)b * HID + c] = (__bf16)tanhf(v);
    }

    // grid barrier across the 16 WGs (release h writes / acquire others')
    __syncthreads();
    if (tid == 0) {
      unsigned g = __hip_atomic_load(&bar[1], __ATOMIC_RELAXED,
                                     __HIP_MEMORY_SCOPE_AGENT);
      if (__hip_atomic_fetch_add(&bar[0], 1u, __ATOMIC_ACQ_REL,
                                 __HIP_MEMORY_SCOPE_AGENT) == 15u) {
        __hip_atomic_store(&bar[0], 0u, __ATOMIC_RELAXED,
                           __HIP_MEMORY_SCOPE_AGENT);
        __hip_atomic_fetch_add(&bar[1], 1u, __ATOMIC_ACQ_REL,
                               __HIP_MEMORY_SCOPE_AGENT);
      } else {
        while (__hip_atomic_load(&bar[1], __ATOMIC_ACQUIRE,
                                 __HIP_MEMORY_SCOPE_AGENT) == g)
          __builtin_amdgcn_s_sleep(1);
      }
    }
    __syncthreads();
  }
}

extern "C" void kernel_launch(void* const* d_in, const int* in_sizes, int n_in,
                              void* d_out, int out_size, void* d_ws, size_t ws_size,
                              hipStream_t stream) {
  (void)in_sizes; (void)n_in; (void)out_size; (void)ws_size;
  const float* X      = (const float*)d_in[0];   // [64,1024,512]
  const float* w_xh   = (const float*)d_in[1];   // [512,512]
  const float* w_hh   = (const float*)d_in[2];   // [512,512]
  const float* bias_h = (const float*)d_in[3];   // [512]
  const float* w_out  = (const float*)d_in[4];   // [512,512] = [out,in]
  const float* b_out  = (const float*)d_in[5];   // [512]
  float* out = (float*)d_out;

  char* ws = (char*)d_ws;
  size_t off = 0;
  float*  xw    = (float*)(ws + off);  off += (size_t)SEQ * BATCH * HID * 4; // 128MB
  __bf16* wxhT  = (__bf16*)(ws + off); off += (size_t)HID * HID * 2;
  __bf16* whhT  = (__bf16*)(ws + off); off += (size_t)HID * HID * 2;
  __bf16* woutB = (__bf16*)(ws + off); off += (size_t)HID * HID * 2;
  __bf16* h0    = (__bf16*)(ws + off); off += (size_t)BATCH * HID * 2;
  __bf16* h1    = (__bf16*)(ws + off); off += (size_t)BATCH * HID * 2;
  unsigned* bar = (unsigned*)(ws + off);

  // weights -> bf16 (+transpose), zero h0 / barrier
  rnn_prep<<<(HID * HID) / 256, 256, 0, stream>>>(w_xh, w_hh, w_out,
                                                  wxhT, whhT, woutB, h0, bar);
  // phase A: xw[t][b][h] = X[b][t][:] . w_xh[:,h] + bias_h
  gemm64_bias<float><<<dim3(SEQ, HID / 64), 256, 0, stream>>>(
      X, (long)HID, (long)SEQ * HID, wxhT, bias_h, xw, (long)BATCH * HID);
  // phase B: 1024 serial steps (final h lands in h0 since SEQ is even)
  rnn_recurrence<<<16, 256, 0, stream>>>(xw, whhT, h0, h1, bar);
  // phase C: out = h_final @ w_out^T + b_out
  gemm64_bias<__bf16><<<dim3(1, HID / 64), 256, 0, stream>>>(
      h0, 0L, (long)HID, woutB, b_out, out, 0L);
}
